// MultiHeadAttentionConvShuffle_50500225466779
// MI455X (gfx1250) — matrix operand order repaired and checked by reading, not compiled, not run on hardware
//
#include <hip/hip_runtime.h>
#include <stdint.h>

// ---------------------------------------------------------------------------
// Problem constants: x (16, 256, 64, 64) f32; wq/wk/wv (256,256,1,1) f32
// ---------------------------------------------------------------------------
#define BATCH   16
#define CHANS   256
#define NPIX    4096          // 64*64
#define HEADS   8
#define DHEAD   32
#define NCHUNK  4             // score-kernel N split

typedef __attribute__((ext_vector_type(16))) __bf16 v16bf;
typedef __attribute__((ext_vector_type(8)))  float  v8f;

static __device__ __forceinline__ uint16_t f2bf(float f) {
    // round-to-nearest-even f32 -> bf16
    uint32_t x = __builtin_bit_cast(uint32_t, f);
    x += 0x7FFFu + ((x >> 16) & 1u);
    return (uint16_t)(x >> 16);
}

// ---------------------------------------------------------------------------
// Kernel 1: fused QKV GEMM.  O(768x4096) = [Wq;Wk;Wv](768x256) * X_b(256x4096)
// 128x64 macro tile / block, 256 threads (8 waves), wave w owns rows
// [16w, 16w+16) => 4 WMMA tiles sharing one A operand per k-step.
// Double-buffered LDS staging (one barrier per k-step) + global_prefetch_b8
// two k-steps ahead on the streaming X rows.
// Q,K written as bf16 (consumed by scores kernel), V written as f32.
// ---------------------------------------------------------------------------
__global__ __launch_bounds__(256) void qkv_wmma(
        const float* __restrict__ x,
        const float* __restrict__ wq,
        const float* __restrict__ wk,
        const float* __restrict__ wv,
        float* __restrict__ Vf32,
        uint16_t* __restrict__ Qbf,
        uint16_t* __restrict__ Kbf)
{
    __shared__ __align__(16) uint16_t ldsA[2][128 * 32];   // [row][k] row-major pairs
    __shared__ __align__(16) uint16_t ldsB[2][64 * 32];    // [n][k]   transposed pairs

    const int tid   = threadIdx.x;
    const int lane  = tid & 31;
    const int wave  = tid >> 5;                 // 0..7
    const int b     = blockIdx.z;
    const int tileM = blockIdx.y * 128;         // 0..640 over stacked [Wq;Wk;Wv]
    const int tileN = blockIdx.x * 64;

    // weight region for this block (uniform: 128-row tiles never straddle 256)
    const float* wsrc;
    int wrow0;
    if (tileM < 256)      { wsrc = wq; wrow0 = tileM; }
    else if (tileM < 512) { wsrc = wk; wrow0 = tileM - 256; }
    else                  { wsrc = wv; wrow0 = tileM - 512; }

    const float* xb = x + (size_t)b * CHANS * NPIX;

    const int  m  = lane & 15;
    const bool hi = lane >= 16;

    v8f acc[4];
    #pragma unroll
    for (int t = 0; t < 4; ++t)
        #pragma unroll
        for (int e = 0; e < 8; ++e) acc[t][e] = 0.0f;

    auto stage = [&](int buf, int k0) {
        // A: 128x32, k contiguous -> coalesced weight-row reads
        for (int idx = tid; idx < 128 * 32; idx += 256) {
            int r = idx >> 5, c = idx & 31;
            ldsA[buf][r * 32 + c] = f2bf(wsrc[(wrow0 + r) * CHANS + (k0 + c)]);
        }
        // B transposed: ldsB[n][ci]; consecutive threads walk n -> coalesced
        for (int idx = tid; idx < 64 * 32; idx += 256) {
            int n = idx & 63, ci = idx >> 6;
            ldsB[buf][n * 32 + ci] = f2bf(xb[(size_t)(k0 + ci) * NPIX + (tileN + n)]);
        }
    };

    auto compute = [&](int buf) {
        const uint32_t* A32 = reinterpret_cast<const uint32_t*>(ldsA[buf]); // [128][16]
        const uint32_t* B32 = reinterpret_cast<const uint32_t*>(ldsB[buf]); // [64][16]
        // A operand: contiguous k order (same as B) — lanes 0-15: K0..15,
        // lanes 16-31: K16..31; dword w -> pair index p.
        union { uint32_t u[8]; v16bf v; } aop;
        #pragma unroll
        for (int w = 0; w < 8; ++w) {
            int p = hi ? 8 + w : w;
            aop.u[w] = A32[(wave * 16 + m) * 16 + p];
        }
        #pragma unroll
        for (int nt = 0; nt < 4; ++nt) {
            // B operand: contiguous k order (same as A) — lanes 0-15: K0..15, lanes 16-31: K16..31
            union { uint32_t u[8]; v16bf v; } bop;
            #pragma unroll
            for (int w = 0; w < 8; ++w)
                bop.u[w] = B32[(nt * 16 + m) * 16 + (hi ? 8 + w : w)];
            acc[nt] = __builtin_amdgcn_wmma_f32_16x16x32_bf16(
                false, aop.v, false, bop.v, (short)0, acc[nt], false, false);
        }
    };

    stage(0, 0);
    __syncthreads();
    for (int kk = 0; kk < 8; ++kk) {
        // prefetch X rows two k-steps ahead (global_prefetch_b8, L2-bound)
        const int pk = (kk + 2) * 32;
        if (pk < CHANS) {
            const int ci = tid >> 3;            // 0..31
            const int n8 = (tid & 7) << 3;      // 0,8,..,56: one line per 8 lanes
            __builtin_prefetch(xb + (size_t)(pk + ci) * NPIX + tileN + n8, 0, 0);
        }
        if (kk + 1 < 8) stage((kk + 1) & 1, (kk + 1) * 32);
        compute(kk & 1);
        __syncthreads();
    }

    // C/D layout: vgpr r -> M = (lane<16 ? r : 8+r), N = lane&15
    #pragma unroll
    for (int nt = 0; nt < 4; ++nt) {
        #pragma unroll
        for (int r = 0; r < 8; ++r) {
            int grow = tileM + wave * 16 + (hi ? 8 + r : r);
            int gcol = tileN + nt * 16 + m;
            float val = acc[nt][r];
            if (grow < 256)
                Qbf[((size_t)b * CHANS + grow) * NPIX + gcol] = f2bf(val);
            else if (grow < 512)
                Kbf[((size_t)b * CHANS + (grow - 256)) * NPIX + gcol] = f2bf(val);
            else
                Vf32[((size_t)b * CHANS + (grow - 512)) * NPIX + gcol] = val;
        }
    }
}

// ---------------------------------------------------------------------------
// Kernel 2: partial scores  S_bh(32x32) += Q_bh(32xNc) * K_bh^T over one of
// NCHUNK N-chunks.  One block per (b,h,chunk); 4 waves -> 16x16 tiles.
// Partials go to per-chunk buffers (fixed-order reduce in softmax kernel ->
// bitwise deterministic).  Q/K rows are bf16-pair contiguous, so WMMA
// operands are direct dword loads.
// ---------------------------------------------------------------------------
__global__ __launch_bounds__(128) void scores_wmma(
        const uint16_t* __restrict__ Qbf,
        const uint16_t* __restrict__ Kbf,
        float* __restrict__ scores)       // NCHUNK consecutive partial buffers
{
    const int tid   = threadIdx.x;
    const int lane  = tid & 31;
    const int wave  = tid >> 5;
    const int b     = blockIdx.x >> 3;
    const int h     = blockIdx.x & 7;
    const int chunk = blockIdx.y;
    const int mi    = wave >> 1;     // 0..1 : d-row half
    const int ni    = wave & 1;      // 0..1 : e-col half

    const uint32_t* Q32 = reinterpret_cast<const uint32_t*>(Qbf);
    const uint32_t* K32 = reinterpret_cast<const uint32_t*>(Kbf);

    const int  m  = lane & 15;
    const bool hi = lane >= 16;

    const size_t qrow = ((size_t)b * CHANS + h * DHEAD + mi * 16 + m) * (NPIX / 2);
    const size_t krow = ((size_t)b * CHANS + h * DHEAD + ni * 16 + m) * (NPIX / 2);

    v8f acc;
    #pragma unroll
    for (int e = 0; e < 8; ++e) acc[e] = 0.0f;

    const int n_lo = chunk * (NPIX / NCHUNK);
    for (int n0 = n_lo; n0 < n_lo + NPIX / NCHUNK; n0 += 32) {
        const int base = n0 >> 1;   // dword offset of this k-window in the row
        union { uint32_t u[8]; v16bf v; } aop, bop;
        #pragma unroll
        for (int w = 0; w < 8; ++w) {
            int pa = hi ? 8 + w : w;
            aop.u[w] = Q32[qrow + base + pa];
            bop.u[w] = K32[krow + base + (hi ? 8 + w : w)];
        }
        acc = __builtin_amdgcn_wmma_f32_16x16x32_bf16(
            false, aop.v, false, bop.v, (short)0, acc, false, false);
    }

    const float scale = 0.35355339059327373f;   // 1/sqrt(8)
    float* dst = scores + (size_t)chunk * (BATCH * HEADS * DHEAD * DHEAD);
    #pragma unroll
    for (int r = 0; r < 8; ++r) {
        int row = mi * 16 + (hi ? 8 + r : r);
        int col = ni * 16 + m;
        dst[(((size_t)b * HEADS + h) * DHEAD + row) * DHEAD + col] = acc[r] * scale;
    }
}

// ---------------------------------------------------------------------------
// Kernel 3: fixed-order chunk reduction + softmax over 32-wide rows
//           -> attn_weights (written to d_out tail)
// ---------------------------------------------------------------------------
__global__ __launch_bounds__(DHEAD) void softmax32(
        const float* __restrict__ scores,
        float* __restrict__ attnW)
{
    const int bh  = blockIdx.x;          // b*8 + h
    const int row = threadIdx.x;         // 0..31
    const size_t base = ((size_t)bh * DHEAD + row) * DHEAD;
    const size_t cstride = (size_t)BATCH * HEADS * DHEAD * DHEAD;

    float v[DHEAD];
    float mx = -3.402823466e+38f;
    #pragma unroll
    for (int e = 0; e < DHEAD; ++e) {
        float s = scores[base + e];
        #pragma unroll
        for (int c = 1; c < NCHUNK; ++c) s += scores[c * cstride + base + e];
        v[e] = s;
        mx = fmaxf(mx, s);
    }
    float s = 0.0f;
    #pragma unroll
    for (int e = 0; e < DHEAD; ++e) { v[e] = expf(v[e] - mx); s += v[e]; }
    const float inv = 1.0f / s;
    #pragma unroll
    for (int e = 0; e < DHEAD; ++e) attnW[base + e] = v[e] * inv;
}

// ---------------------------------------------------------------------------
// Kernel 4: attn = A(32x32) * V_bh(32x4096), channel shuffle (c = d*8 + h),
// residual add with V.  Memory-bound streaming kernel; A cached in LDS.
// ---------------------------------------------------------------------------
__global__ __launch_bounds__(256) void attn_apply(
        const float* __restrict__ attnW,
        const float* __restrict__ Vf32,
        float* __restrict__ out)
{
    __shared__ float Aw[DHEAD * DHEAD];
    const int tid = threadIdx.x;        // 256
    const int h   = blockIdx.y;
    const int b   = blockIdx.z;

    const float* Abh = attnW + ((size_t)b * HEADS + h) * DHEAD * DHEAD;
    for (int idx = tid; idx < DHEAD * DHEAD; idx += 256) Aw[idx] = Abh[idx];
    __syncthreads();

    const int n = blockIdx.x * 256 + tid;
    const float* Vb = Vf32 + (size_t)b * CHANS * NPIX;
    float*       Ob = out  + (size_t)b * CHANS * NPIX;

    float vv[DHEAD];
    #pragma unroll
    for (int e = 0; e < DHEAD; ++e)
        vv[e] = Vb[(size_t)(h * DHEAD + e) * NPIX + n];   // coalesced across lanes

    #pragma unroll 4
    for (int d = 0; d < DHEAD; ++d) {
        const int c = d * HEADS + h;                      // shuffled channel
        float s = Vb[(size_t)c * NPIX + n];               // residual
        #pragma unroll
        for (int e = 0; e < DHEAD; ++e) s = fmaf(Aw[d * DHEAD + e], vv[e], s);
        Ob[(size_t)c * NPIX + n] = s;
    }
}

// ---------------------------------------------------------------------------
// Launch
// ---------------------------------------------------------------------------
extern "C" void kernel_launch(void* const* d_in, const int* in_sizes, int n_in,
                              void* d_out, int out_size, void* d_ws, size_t ws_size,
                              hipStream_t stream)
{
    (void)in_sizes; (void)n_in; (void)out_size; (void)ws_size;

    const float* x  = (const float*)d_in[0];
    const float* wq = (const float*)d_in[1];
    const float* wk = (const float*)d_in[2];
    const float* wv = (const float*)d_in[3];

    float* out      = (float*)d_out;
    float* attn_out = out + (size_t)BATCH * CHANS * NPIX;   // tuple tail

    char* ws = (char*)d_ws;
    float*    Vf32   = (float*)(ws);                                   // 64 MB
    uint16_t* Qbf    = (uint16_t*)(ws + (size_t)67108864);             // 32 MB
    uint16_t* Kbf    = (uint16_t*)(ws + (size_t)100663296);            // 32 MB
    float*    scores = (float*)(ws + (size_t)134217728);               // 4 x 512 KB

    // 1) QKV GEMM: grid (Ntiles=64, Mtiles=6 over 768 stacked rows, b=16)
    qkv_wmma<<<dim3(NPIX / 64, 768 / 128, BATCH), 256, 0, stream>>>(
        x, wq, wk, wv, Vf32, Qbf, Kbf);

    // 2) channel-attention partial scores, one block per (b,h,chunk)
    scores_wmma<<<dim3(BATCH * HEADS, NCHUNK), 128, 0, stream>>>(Qbf, Kbf, scores);

    // 3) deterministic chunk reduce + softmax -> attn_weights output
    softmax32<<<dim3(BATCH * HEADS), DHEAD, 0, stream>>>(scores, attn_out);

    // 4) apply attention + shuffle + residual
    attn_apply<<<dim3(NPIX / 256, HEADS, BATCH), 256, 0, stream>>>(
        attn_out, Vf32, out);
}
